// PiKVCompressor_60344290509596
// MI455X (gfx1250) — matrix operand order repaired and checked by reading, not compiled, not run on hardware
//
#include <hip/hip_runtime.h>

#define NTOK 16384
#define HDIM 1024
#define T0   3276          // level-0 size: int(16384*0.2)
#define T1   5373          // T0 + int(13108*0.16)
#define CNT1 2097
#define CNT2 11011

typedef __attribute__((ext_vector_type(16))) _Float16 v16h;
typedef __attribute__((ext_vector_type(8)))  _Float16 v8h;
typedef __attribute__((ext_vector_type(2)))  __fp16   h2;   // cvt_pkrtz return type
typedef __attribute__((ext_vector_type(8)))  float    v8f;

union V8H { h2 p[4]; v8h v; };

// ---------------------------------------------------------------------------
// Rank kernel: order[rank(i)] = i, rank = stable-descending-argsort position.
// ---------------------------------------------------------------------------
__global__ __launch_bounds__(256) void rank_kernel(const float* __restrict__ imp,
                                                   int* __restrict__ order) {
  __shared__ float s[NTOK];
  const int t = threadIdx.x;
  const float4* g4 = (const float4*)imp;
  float4* s4 = (float4*)s;
  for (int i = t; i < NTOK / 4; i += 256) s4[i] = g4[i];
  __syncthreads();
  const int i = blockIdx.x * 256 + t;
  const float x = s[i];
  int cnt = 0;
  for (int j4 = 0; j4 < NTOK / 4; ++j4) {
    const float4 q = s4[j4];
    const int j = j4 * 4;
    cnt += (q.x > x) || (q.x == x && (j + 0) < i);
    cnt += (q.y > x) || (q.y == x && (j + 1) < i);
    cnt += (q.z > x) || (q.z == x && (j + 2) < i);
    cnt += (q.w > x) || (q.w == x && (j + 3) < i);
  }
  order[cnt] = i;
}

// ---------------------------------------------------------------------------
// Level-0: straight copy of the 3276 highest-importance rows (keys & values).
// ---------------------------------------------------------------------------
__global__ __launch_bounds__(256) void copy_l0(const float* __restrict__ keys,
                                               const float* __restrict__ values,
                                               const int* __restrict__ order,
                                               float* __restrict__ out) {
  const int token  = order[blockIdx.x];
  const int tensor = blockIdx.y;
  const float* src = (tensor ? values : keys) + (long)token * HDIM;
  float* dst = out + (long)tensor * NTOK * HDIM + (long)token * HDIM;
  ((float4*)dst)[threadIdx.x] = ((const float4*)src)[threadIdx.x];
}

// ---------------------------------------------------------------------------
__device__ __forceinline__ const float* src_row(const float* s0, const float* s1,
    const int* gl, long m, int count, int ld) {
  if (gl) {
    const int tn  = m >= count;
    const int tok = gl[m - (long)tn * count];
    return (tn ? s1 : s0) + (long)tok * ld;
  }
  return s0 + m * (long)ld;
}
__device__ __forceinline__ float* dst_row(float* o0, float* o1,
    const int* sl, long m, int count, int ld) {
  if (sl) {
    const int tn  = m >= count;
    const int tok = sl[m - (long)tn * count];
    return (tn ? o1 : o0) + (long)tok * ld;
  }
  return o0 + m * (long)ld;
}

// ---------------------------------------------------------------------------
// OUT = relu(X @ W + bias).  Block tile 256x64, 8 waves (32x64 strip each),
// K-step 32, 8x v_wmma_f32_16x16x32_f16 per wave per step with B fragments
// reused across both M-halves.  K and N are COMPILE-TIME so B prefetch is a
// single running pointer + immediate offsets (i*N*4 fits the 24-bit ioffset)
// and trip counts are constants.  Software-pipelined global->reg prefetch,
// packed f16 conversion, b128 LDS staging.  Out-of-range rows/cols CLAMPED
// at load (garbage masked at output store): no divergent guards in the loop.
// The K%32 tail (K=204) is a peeled, masked step in the TAIL instantiation.
// ---------------------------------------------------------------------------
template <int K, int N, bool TAIL = false>
__global__ __launch_bounds__(256) void gemm_relu_wmma(
    const float* __restrict__ s0, const float* __restrict__ s1,
    const int* __restrict__ glist, int src_ld,
    const float* __restrict__ W, const float* __restrict__ bias,
    int Mtot, int count,
    float* __restrict__ o0, float* __restrict__ o1,
    const int* __restrict__ slist, int out_ld) {
  __shared__ _Float16 lA[256 * 40];   // A tile: 256 rows x 32 K (f16)
  __shared__ _Float16 lB[64 * 40];    // B tile transposed: 64 cols x 32 K (f16)

  const int t  = threadIdx.x;
  const int n0 = blockIdx.x * 64;
  const long m0 = (long)blockIdx.y * 256;

  // A staging role: one full 32-element row per thread (row clamped)
  const int ar = t;
  const long am = m0 + ar;
  const long amc = (am < Mtot) ? am : (long)(Mtot - 1);
  const float* xp = src_row(s0, s1, glist, amc, count, src_ld);  // bumped +32/step

  // B staging role: 1 column, 8 consecutive K each (column clamped)
  const int bn  = t & 63;             // column within tile
  const int bkc = (t >> 6) * 8;       // k chunk: 0,8,16,24
  const int bcolr = n0 + bn;
  const float* wpk = W + ((bcolr < N) ? bcolr : (N - 1)) + (long)bkc * N;

  // wave roles
  const int wv   = t >> 5;
  const int lane = t & 31;
  const int lr   = lane & 15;
  const int hi   = lane >> 4;

  v8f acc0[4] = {};                   // rows wv*32 + [0,16)
  v8f acc1[4] = {};                   // rows wv*32 + [16,32)
  float4 fa[8];                       // prefetched A row (32 f32)
  float  fb[8];                       // prefetched B chunk (8 f32)

  auto loadA = [&]() {                // src_ld >= ceil32(K): always in-bounds
    #pragma unroll
    for (int q = 0; q < 8; ++q) fa[q] = *(const float4*)(xp + q * 4);
  };
  auto loadB_fast = [&]() {           // base pointer + compile-time offsets
    #pragma unroll
    for (int i = 0; i < 8; ++i) fb[i] = wpk[i * N];
  };
  auto loadB_tail = [&](int k0) {     // zero-masked on k >= K
    #pragma unroll
    for (int i = 0; i < 8; ++i)
      fb[i] = (k0 + bkc + i < K) ? wpk[i * N] : 0.f;
  };
  auto commitB = [&]() {
    V8H b;
    #pragma unroll
    for (int i = 0; i < 4; ++i)
      b.p[i] = __builtin_amdgcn_cvt_pkrtz(fb[i * 2], fb[i * 2 + 1]);
    *(v8h*)&lB[bn * 40 + bkc] = b.v;     // 80*bn + {0,16,32,48}: 16B aligned
  };
  auto commitA_fast = [&]() {
    V8H a[4];
    #pragma unroll
    for (int q = 0; q < 4; ++q) {
      a[q].p[0] = __builtin_amdgcn_cvt_pkrtz(fa[2 * q].x, fa[2 * q].y);
      a[q].p[1] = __builtin_amdgcn_cvt_pkrtz(fa[2 * q].z, fa[2 * q].w);
      a[q].p[2] = __builtin_amdgcn_cvt_pkrtz(fa[2 * q + 1].x, fa[2 * q + 1].y);
      a[q].p[3] = __builtin_amdgcn_cvt_pkrtz(fa[2 * q + 1].z, fa[2 * q + 1].w);
    }
    v8h* dA = (v8h*)&lA[ar * 40];        // 80*ar: 16B aligned, 4x b128
    #pragma unroll
    for (int q = 0; q < 4; ++q) dA[q] = a[q].v;
  };
  auto commitA_tail = [&](int k0) {      // zero-mask elements with k >= K
    float e[32];
    #pragma unroll
    for (int q = 0; q < 8; ++q) {
      e[q * 4 + 0] = fa[q].x; e[q * 4 + 1] = fa[q].y;
      e[q * 4 + 2] = fa[q].z; e[q * 4 + 3] = fa[q].w;
    }
    #pragma unroll
    for (int q = 0; q < 32; ++q)
      if (k0 + q >= K) e[q] = 0.f;
    V8H a[4];
    #pragma unroll
    for (int q = 0; q < 4; ++q)
      #pragma unroll
      for (int p = 0; p < 4; ++p)
        a[q].p[p] = __builtin_amdgcn_cvt_pkrtz(e[q * 8 + p * 2], e[q * 8 + p * 2 + 1]);
    v8h* dA = (v8h*)&lA[ar * 40];
    #pragma unroll
    for (int q = 0; q < 4; ++q) dA[q] = a[q].v;
  };
  auto do_mma = [&]() {
    union { v16h v; v8h h[2]; } af0, af1;
    const _Float16* ap0 = &lA[(wv * 32 + lr) * 40 + hi * 8];
    af0.h[0] = *(const v8h*)ap0;
    af0.h[1] = *(const v8h*)(ap0 + 16);
    const _Float16* ap1 = ap0 + 16 * 40;
    af1.h[0] = *(const v8h*)ap1;
    af1.h[1] = *(const v8h*)(ap1 + 16);
    #pragma unroll
    for (int j = 0; j < 4; ++j) {
      union { v16h v; v8h h[2]; } bf;
      const _Float16* bp = &lB[(j * 16 + lr) * 40 + hi * 8];
      bf.h[0] = *(const v8h*)bp;
      bf.h[1] = *(const v8h*)(bp + 16);
      acc0[j] = __builtin_amdgcn_wmma_f32_16x16x32_f16(
          false, af0.v, false, bf.v, (short)0, acc0[j], false, false);
      acc1[j] = __builtin_amdgcn_wmma_f32_16x16x32_f16(
          false, af1.v, false, bf.v, (short)0, acc1[j], false, false);
    }
  };

  constexpr int nfull = TAIL ? (K >> 5) : ((K + 31) >> 5);

  loadA();
  loadB_fast();
  for (int ks = 0; ks < nfull; ++ks) {
    commitA_fast();
    commitB();
    __syncthreads();
    xp  += 32;                        // pointer bumps (one 64-bit add each)
    wpk += 32 * N;
    if (ks + 1 < nfull) {             // prefetch next step during WMMA section
      loadA();
      loadB_fast();
    } else if (TAIL) {
      loadA();
      loadB_tail((ks + 1) << 5);
    }
    do_mma();
    __syncthreads();
  }
  if (TAIL) {                         // peeled, masked final K-step
    constexpr int k0 = nfull << 5;
    commitA_tail(k0);
    commitB();                        // fb already zero-masked at load
    __syncthreads();
    do_mma();
  }

  // ---- epilogue: bias + ReLU, dense or scattered f32 store ----
  float bc[4];
  #pragma unroll
  for (int j = 0; j < 4; ++j) {
    const int col = n0 + j * 16 + lr;
    bc[j] = (col < N) ? bias[col] : 0.f;
  }
  #pragma unroll
  for (int h = 0; h < 2; ++h) {
    #pragma unroll
    for (int r = 0; r < 8; ++r) {
      const long m = m0 + wv * 32 + h * 16 + hi * 8 + r;  // lanes>=16 -> M+8
      if (m < Mtot) {
        float* orow = dst_row(o0, o1, slist, m, count, out_ld);
        #pragma unroll
        for (int j = 0; j < 4; ++j) {
          const int col = n0 + j * 16 + lr;
          if (col < N) {
            const float v = (h ? acc1[j][r] : acc0[j][r]) + bc[j];
            orow[col] = v > 0.f ? v : 0.f;
          }
        }
      }
    }
  }
}

// ---------------------------------------------------------------------------
extern "C" void kernel_launch(void* const* d_in, const int* in_sizes, int n_in,
                              void* d_out, int out_size, void* d_ws, size_t ws_size,
                              hipStream_t stream) {
  const float* keys   = (const float*)d_in[0];
  const float* values = (const float*)d_in[1];
  const float* imp    = (const float*)d_in[2];
  const float* We0 = (const float*)d_in[3];  const float* be0 = (const float*)d_in[4];
  const float* We1 = (const float*)d_in[5];  const float* be1 = (const float*)d_in[6];
  const float* Wd0 = (const float*)d_in[7];  const float* bd0 = (const float*)d_in[8];
  const float* Wd1 = (const float*)d_in[9];  const float* bd1 = (const float*)d_in[10];

  float* out_k = (float*)d_out;
  float* out_v = out_k + (size_t)NTOK * HDIM;

  // workspace: order[16384] | bufA (22022x512 f32) | bufB (22022x224 f32) ~ 62 MiB
  int*   order = (int*)d_ws;
  float* bufA  = (float*)((char*)d_ws + NTOK * sizeof(int));
  float* bufB  = bufA + (size_t)(2 * CNT2) * 512;

  rank_kernel<<<NTOK / 256, 256, 0, stream>>>(imp, order);
  copy_l0<<<dim3(T0, 2), 256, 0, stream>>>(keys, values, order, (float*)d_out);

  const int* l1 = order + T0;
  const int* l2 = order + T1;

  // ---- level 1: 1024 -> 512 -> 1024 (2097 tokens, k & v folded into M) ----
  {
    const int M = 2 * CNT1;
    const int yb = (M + 255) / 256;
    gemm_relu_wmma<1024, 512><<<dim3(8, yb), 256, 0, stream>>>(
        keys, values, l1, HDIM, We0, be0, M, CNT1,
        bufA, nullptr, nullptr, 512);
    gemm_relu_wmma<512, 1024><<<dim3(16, yb), 256, 0, stream>>>(
        bufA, nullptr, nullptr, 512, Wd0, bd0, M, CNT1,
        out_k, out_v, l1, HDIM);
  }
  // ---- level 2: 1024 -> 512 -> 204 -> 512 -> 1024 (11011 tokens) ----
  {
    const int M = 2 * CNT2;
    const int yb = (M + 255) / 256;
    gemm_relu_wmma<1024, 512><<<dim3(8, yb), 256, 0, stream>>>(
        keys, values, l2, HDIM, We0, be0, M, CNT2,
        bufA, nullptr, nullptr, 512);
    gemm_relu_wmma<512, 204><<<dim3(4, yb), 256, 0, stream>>>(
        bufA, nullptr, nullptr, 512, We1, be1, M, CNT2,
        bufB, nullptr, nullptr, 224);
    gemm_relu_wmma<204, 512, true><<<dim3(8, yb), 256, 0, stream>>>(
        bufB, nullptr, nullptr, 224, Wd1, bd1, M, CNT2,
        bufA, nullptr, nullptr, 512);
    gemm_relu_wmma<512, 1024><<<dim3(16, yb), 256, 0, stream>>>(
        bufA, nullptr, nullptr, 512, Wd0, bd0, M, CNT2,
        out_k, out_v, l2, HDIM);
  }
}
